// QuantLLaMAAttention_35021163331727
// MI455X (gfx1250) — matrix operand order repaired and checked by reading, not compiled, not run on hardware
//
#include <hip/hip_runtime.h>
#include <hip/hip_bf16.h>
#include <stdint.h>

#define NUMHEADS 32
#define NUMKVHEADS 8
#define HEADDIM 128
#define SEQ 2048
#define BATCH 2
#define HID 4096
#define KVD 1024
#define ROWS (BATCH*SEQ) /* 4096 */

typedef __attribute__((ext_vector_type(16))) __bf16 v16bf;
typedef __attribute__((ext_vector_type(8)))  float  v8f;
typedef __attribute__((ext_vector_type(4)))  unsigned int v4u;
typedef __attribute__((ext_vector_type(8)))  int v8i;
typedef __attribute__((ext_vector_type(4)))  int v4i;

union frag16 { v16bf v; uint32_t u[8]; };

__device__ __forceinline__ uint16_t f2bf(float f) {
  union { float f; uint32_t u; } x{f};
  uint32_t r = x.u + 0x7fffu + ((x.u >> 16) & 1u);   // round-to-nearest-even
  return (uint16_t)(r >> 16);
}
__device__ __forceinline__ float bf2f(uint16_t b) {
  union { uint32_t u; float f; } x{(uint32_t)b << 16};
  return x.f;
}

// ---------------------------------------------------------------- TDM helpers
#if __has_builtin(__builtin_amdgcn_tensor_load_to_lds)
#define HAVE_TDM 1
#else
#define HAVE_TDM 0
#endif

#if HAVE_TDM
// 2D tile of bf16 (2-byte) elements: tile_d0 contiguous elems, tile_d1 rows,
// row stride = stride_elems. D# bit layout per CDNA5 ISA ch.8.
__device__ __forceinline__ void tdm_load_2d_bf16(uint32_t lds_addr, const void* gptr,
                                                 uint32_t tile_d0, uint32_t tile_d1,
                                                 uint32_t stride_elems) {
  uint64_t ga = (uint64_t)(uintptr_t)gptr;
  v4u g0;
  g0[0] = 1u;                                                   // count=1 (user D#)
  g0[1] = lds_addr;                                             // lds_addr[31:0]
  g0[2] = (uint32_t)ga;                                         // global_addr[31:0]
  g0[3] = (uint32_t)((ga >> 32) & 0x01FFFFFFull) | (2u << 30);  // addr[56:32] | type=2
  v8i g1;
  g1[0] = (int)(1u << 16);                                      // data_size=1 (2 bytes)
  g1[1] = (int)((tile_d0 & 0xFFFFu) << 16);                     // tensor_dim0[15:0]
  g1[2] = (int)(((tile_d0 >> 16) & 0xFFFFu) |
                ((tile_d1 & 0xFFFFu) << 16));                   // dim0 hi | tensor_dim1 lo
  g1[3] = (int)(((tile_d1 >> 16) & 0xFFFFu) |
                ((tile_d0 & 0xFFFFu) << 16));                   // dim1 hi | tile_dim0
  g1[4] = (int)(tile_d1 & 0xFFFFu);                             // tile_dim1 (tile_dim2=0)
  g1[5] = (int)stride_elems;                                    // tensor_dim0_stride[31:0]
  g1[6] = 0;
  g1[7] = 0;
  v4i z = {0, 0, 0, 0};
#if defined(__clang_major__) && (__clang_major__ >= 23)
  v8i z8 = {0, 0, 0, 0, 0, 0, 0, 0};
  __builtin_amdgcn_tensor_load_to_lds(g0, g1, z, z, z8, 0);
#else
  __builtin_amdgcn_tensor_load_to_lds(g0, g1, z, z, 0);
#endif
}
__device__ __forceinline__ void tdm_wait0() {
#if __has_builtin(__builtin_amdgcn_s_wait_tensorcnt)
  __builtin_amdgcn_s_wait_tensorcnt(0);
#else
  asm volatile("s_wait_tensorcnt 0x0" ::: "memory");
#endif
}
#endif

// ---------------------------------------------------------------- fp32 -> bf16
__global__ void k_cvt(const float* __restrict__ src, uint16_t* __restrict__ dst, int n4) {
  int i = blockIdx.x * blockDim.x + threadIdx.x;
  if (i >= n4) return;
  float4 f = ((const float4*)src)[i];
  uint32_t u0 = (uint32_t)f2bf(f.x) | ((uint32_t)f2bf(f.y) << 16);
  uint32_t u1 = (uint32_t)f2bf(f.z) | ((uint32_t)f2bf(f.w) << 16);
  ((uint2*)dst)[i] = make_uint2(u0, u1);
}

// ------------------------------------------------- GEMM: C = A(bf16) * W(f32)^T
// A: [M][K] bf16 row-major. W: [N][K] f32 row-major (torch Linear weight).
// Block tile 128x64, 8 waves (4x2), each wave a 32x32 tile = 4 WMMAs per K-step.
// Double-buffered LDS: TDM prefetch of A tile k+1 + W-tile global loads overlap
// the WMMAs on tile k; s_wait_tensorcnt only just before the barrier.
template <bool OUT_BF16>
__global__ __launch_bounds__(256)
void k_gemm_bf16w(const uint16_t* __restrict__ A, const float* __restrict__ W,
                  void* __restrict__ Cout, int M, int N, int K) {
  __shared__ uint16_t As[2][128 * 32];
  __shared__ uint16_t Bs[2][64 * 32];
  const int tid = threadIdx.x;
  const int lane = tid & 31, wave = tid >> 5;
  const int wm = wave >> 1, wn = wave & 1;
  const int half = lane >> 4, l16 = lane & 15;
  const int r0 = blockIdx.x * 128, c0 = blockIdx.y * 64;
  const int bn = (tid * 8) >> 5, bk = (tid * 8) & 31;   // W staging coords
#if !HAVE_TDM
  const int am = tid >> 1, ak = (tid & 1) * 16;          // A staging coords
#endif
  v8f acc[2][2];
#pragma unroll
  for (int i = 0; i < 2; i++)
#pragma unroll
    for (int j = 0; j < 2; j++) acc[i][j] = (v8f){};

  // ---- prologue: stage tile 0 into buffer 0
#if HAVE_TDM
  if (wave == 0)
    tdm_load_2d_bf16((uint32_t)(size_t)As[0], A + (size_t)r0 * K, 32, 128, (uint32_t)K);
#else
  {
    const uint16_t* s = A + (size_t)(r0 + am) * K + ak;
    *(uint4*)(As[0] + am * 32 + ak) = *(const uint4*)s;
    *(uint4*)(As[0] + am * 32 + ak + 8) = *(const uint4*)(s + 8);
  }
#endif
  {
    const float* s = W + (size_t)(c0 + bn) * K + bk;
    float4 f0 = *(const float4*)s, f1 = *(const float4*)(s + 4);
    uint16_t* p = Bs[0] + bn * 32 + bk;
    p[0] = f2bf(f0.x); p[1] = f2bf(f0.y); p[2] = f2bf(f0.z); p[3] = f2bf(f0.w);
    p[4] = f2bf(f1.x); p[5] = f2bf(f1.y); p[6] = f2bf(f1.z); p[7] = f2bf(f1.w);
  }
#if HAVE_TDM
  if (wave == 0) tdm_wait0();
#endif
  __syncthreads();

  for (int k0 = 0; k0 < K; k0 += 32) {
    const int buf = (k0 >> 5) & 1, nbuf = buf ^ 1;
    const bool havenext = (k0 + 32) < K;
    float4 f0, f1;                    // next W tile held in regs through compute
#if !HAVE_TDM
    uint4 a0, a1;
#endif
    if (havenext) {
#if HAVE_TDM
      if (wave == 0)                  // async DMA of next 128x32 A tile
        tdm_load_2d_bf16((uint32_t)(size_t)As[nbuf],
                         A + (size_t)r0 * K + k0 + 32, 32, 128, (uint32_t)K);
#else
      {
        const uint16_t* s = A + (size_t)(r0 + am) * K + k0 + 32 + ak;
        a0 = *(const uint4*)s; a1 = *(const uint4*)(s + 8);
      }
#endif
      const float* s = W + (size_t)(c0 + bn) * K + k0 + 32 + bk;
      f0 = *(const float4*)s; f1 = *(const float4*)(s + 4);
    }

    // ---- compute on current buffer
    frag16 a[2], b[2];
#pragma unroll
    for (int ms = 0; ms < 2; ms++)
#pragma unroll
      for (int i = 0; i < 8; i++) { // A-frag: m=l16, K pairs per ISA layout
        int kb = (i < 4 ? 2 * i : 16 + 2 * (i - 4)) + 8 * half;
        a[ms].u[i] = *(const uint32_t*)(As[buf] + (wm * 32 + ms * 16 + l16) * 32 + kb);
      }
#pragma unroll
    for (int ns = 0; ns < 2; ns++)
#pragma unroll
      for (int i = 0; i < 8; i++) { // B-frag: n=l16, K pairs per ISA layout
        int kb = (i < 4 ? 2 * i : 16 + 2 * (i - 4)) + 8 * half;
        b[ns].u[i] = *(const uint32_t*)(Bs[buf] + (wn * 32 + ns * 16 + l16) * 32 + kb);
      }
#pragma unroll
    for (int ms = 0; ms < 2; ms++)
#pragma unroll
      for (int ns = 0; ns < 2; ns++)
        acc[ms][ns] = __builtin_amdgcn_wmma_f32_16x16x32_bf16(
            false, a[ms].v, false, b[ns].v, (short)0, acc[ms][ns], false, false);

    // ---- commit next tile to the other buffer
    if (havenext) {
      uint16_t* p = Bs[nbuf] + bn * 32 + bk;
      p[0] = f2bf(f0.x); p[1] = f2bf(f0.y); p[2] = f2bf(f0.z); p[3] = f2bf(f0.w);
      p[4] = f2bf(f1.x); p[5] = f2bf(f1.y); p[6] = f2bf(f1.z); p[7] = f2bf(f1.w);
#if !HAVE_TDM
      *(uint4*)(As[nbuf] + am * 32 + ak) = a0;
      *(uint4*)(As[nbuf] + am * 32 + ak + 8) = a1;
#else
      if (wave == 0) tdm_wait0();
#endif
    }
    __syncthreads();
  }
#pragma unroll
  for (int ms = 0; ms < 2; ms++)
#pragma unroll
    for (int ns = 0; ns < 2; ns++)
#pragma unroll
      for (int i = 0; i < 8; i++) { // C-frag: row = i + 8*half, col = l16
        int row = r0 + wm * 32 + ms * 16 + i + 8 * half;
        int col = c0 + wn * 32 + ns * 16 + l16;
        if (OUT_BF16) ((uint16_t*)Cout)[(size_t)row * N + col] = f2bf(acc[ms][ns][i]);
        else          ((float*)Cout)[(size_t)row * N + col] = acc[ms][ns][i];
      }
}

// ---------------------------------------------------------------- RoPE (bf16, in place)
__global__ void k_rope(uint16_t* __restrict__ X, int nheads, int ld) {
  int id = blockIdx.x * blockDim.x + threadIdx.x;
  int ppr = nheads * 64;
  int row = id / ppr;
  if (row >= ROWS) return;
  int rem = id - row * ppr;
  int hh = rem >> 6, d = rem & 63;
  int pos = row & (SEQ - 1);
  float inv = __expf((float)d * (-2.0f / 128.0f) * 9.210340371976184f); // theta^-2d/D
  float ang = (float)pos * inv;
  float s, c; __sincosf(ang, &s, &c);
  size_t base = (size_t)row * ld + hh * 128 + d;
  float x1 = bf2f(X[base]), x2 = bf2f(X[base + 64]);
  X[base]      = f2bf(x1 * c - x2 * s);
  X[base + 64] = f2bf(x2 * c + x1 * s);
}

// ------------------------------------------------- Flash attention (causal, GQA)
// grid: (SEQ/128, BATCH*NUMHEADS), 256 threads; each wave owns 16 query rows.
// Double-buffered K (TDM) / V (manual transpose) tiles; next tile's DMA and
// global loads are issued before the compute phase and committed after it.
__global__ __launch_bounds__(256)
void k_flash_attn(const uint16_t* __restrict__ Q,   // [ROWS][HID]
                  const uint16_t* __restrict__ Kg,  // [ROWS][KVD]
                  const uint16_t* __restrict__ Vg,  // [ROWS][KVD]
                  uint16_t* __restrict__ Og) {      // [ROWS][HID]
  __shared__ uint16_t Ks[2][32 * 128];   // [key][d]
  __shared__ uint16_t Vs[2][128 * 32];   // [d][key]  (transposed for PV B-frags)
  __shared__ uint16_t Ps[8 * 16 * 32];   // per-wave P re-layout scratch
  const int tid = threadIdx.x, lane = tid & 31, wave = tid >> 5;
  const int half = lane >> 4, l16 = lane & 15;
  const int bh = blockIdx.y, b = bh >> 5, h = bh & 31;
  const int kvh = h >> 2;                 // repeat_interleave(groups=4)
  const int q0 = blockIdx.x * 128;
  const int qw = q0 + wave * 16;
  const int skey = tid >> 3, sd = (tid & 7) * 16;   // staging coords
  const float SCALE = 0.08838834764831845f;         // 1/sqrt(128)

  frag16 qf[4];
  {
    size_t rowbase = ((size_t)(b * SEQ + qw + l16)) * HID + h * HEADDIM;
#pragma unroll
    for (int c = 0; c < 4; c++)
#pragma unroll
      for (int i = 0; i < 8; i++) {
        int kb = (i < 4 ? 2 * i : 16 + 2 * (i - 4)) + 8 * half;
        qf[c].u[i] = *(const uint32_t*)(Q + rowbase + c * 32 + kb);
      }
  }

  float m_i[8], l_i[8];
  v8f o[8];
#pragma unroll
  for (int i = 0; i < 8; i++) { m_i[i] = -1e30f; l_i[i] = 0.f; }
#pragma unroll
  for (int t = 0; t < 8; t++) o[t] = (v8f){};

  // ---- prologue: stage tile j0=0 into buffer 0
  {
    size_t src0 = ((size_t)(b * SEQ)) * KVD + kvh * HEADDIM;
#if HAVE_TDM
    if (wave == 0)
      tdm_load_2d_bf16((uint32_t)(size_t)Ks[0], Kg + src0, 128, 32, (uint32_t)KVD);
#endif
    size_t src = src0 + (size_t)skey * KVD + sd;
    union { uint4 q[2]; uint16_t s[16]; } tv;
    tv.q[0] = *(const uint4*)(Vg + src);
    tv.q[1] = *(const uint4*)(Vg + src + 8);
#pragma unroll
    for (int e = 0; e < 16; e++) Vs[0][(sd + e) * 32 + skey] = tv.s[e];
#if !HAVE_TDM
    *(uint4*)(Ks[0] + skey * 128 + sd) = *(const uint4*)(Kg + src);
    *(uint4*)(Ks[0] + skey * 128 + sd + 8) = *(const uint4*)(Kg + src + 8);
#endif
#if HAVE_TDM
    if (wave == 0) tdm_wait0();
#endif
    __syncthreads();
  }

  for (int j0 = 0; j0 <= q0 + 127; j0 += 32) {
    const int buf = (j0 >> 5) & 1, nbuf = buf ^ 1;
    const bool havenext = (j0 + 32) <= (q0 + 127);
    union { uint4 q[2]; uint16_t s[16]; } tv;   // next V tile held through compute
#if !HAVE_TDM
    uint4 ka0, ka1;
#endif
    if (havenext) {
      size_t src0 = ((size_t)(b * SEQ + j0 + 32)) * KVD + kvh * HEADDIM;
#if HAVE_TDM
      if (wave == 0)   // async next 32x128 bf16 K tile -> LDS
        tdm_load_2d_bf16((uint32_t)(size_t)Ks[nbuf], Kg + src0, 128, 32, (uint32_t)KVD);
#endif
      size_t src = src0 + (size_t)skey * KVD + sd;
      tv.q[0] = *(const uint4*)(Vg + src);
      tv.q[1] = *(const uint4*)(Vg + src + 8);
#if !HAVE_TDM
      ka0 = *(const uint4*)(Kg + src);
      ka1 = *(const uint4*)(Kg + src + 8);
#endif
    }

    // S = Q * K^T for two 16-key tiles (K-dim = d = 128, 4 WMMA steps each)
    v8f s0 = (v8f){}, s1 = (v8f){};
#pragma unroll
    for (int c = 0; c < 4; c++) {
      frag16 kf0, kf1;
#pragma unroll
      for (int i = 0; i < 8; i++) {
        int kb = (i < 4 ? 2 * i : 16 + 2 * (i - 4)) + 8 * half;
        kf0.u[i] = *(const uint32_t*)(Ks[buf] + l16 * 128 + c * 32 + kb);
        kf1.u[i] = *(const uint32_t*)(Ks[buf] + (16 + l16) * 128 + c * 32 + kb);
      }
      s0 = __builtin_amdgcn_wmma_f32_16x16x32_bf16(false, qf[c].v, false, kf0.v,
                                                   (short)0, s0, false, false);
      s1 = __builtin_amdgcn_wmma_f32_16x16x32_bf16(false, qf[c].v, false, kf1.v,
                                                   (short)0, s1, false, false);
    }

    // online softmax (fp32), row reductions across 16-lane groups
    int kg0 = j0 + l16, kg1 = j0 + 16 + l16;
#pragma unroll
    for (int i = 0; i < 8; i++) {
      int qg = qw + i + 8 * half;
      float a0 = (kg0 <= qg) ? s0[i] * SCALE : -1e30f;
      float a1 = (kg1 <= qg) ? s1[i] * SCALE : -1e30f;
      float rm = fmaxf(a0, a1);
      rm = fmaxf(rm, __shfl_xor(rm, 1));
      rm = fmaxf(rm, __shfl_xor(rm, 2));
      rm = fmaxf(rm, __shfl_xor(rm, 4));
      rm = fmaxf(rm, __shfl_xor(rm, 8));
      float mn = fmaxf(m_i[i], rm);
      float al = __expf(m_i[i] - mn);
      float p0 = __expf(a0 - mn);
      float p1 = __expf(a1 - mn);
      float rs = p0 + p1;
      rs += __shfl_xor(rs, 1);
      rs += __shfl_xor(rs, 2);
      rs += __shfl_xor(rs, 4);
      rs += __shfl_xor(rs, 8);
      l_i[i] = al * l_i[i] + rs;
      m_i[i] = mn;
      s0[i] = p0; s1[i] = p1;
#pragma unroll
      for (int t = 0; t < 8; t++) o[t][i] *= al;
    }

    // C-frag -> A-frag re-layout of P via per-wave LDS scratch
    uint16_t* pw = Ps + wave * (16 * 32);
#pragma unroll
    for (int i = 0; i < 8; i++) {
      int mrow = i + 8 * half;
      pw[mrow * 32 + l16]      = f2bf(s0[i]);
      pw[mrow * 32 + 16 + l16] = f2bf(s1[i]);
    }
    asm volatile("s_wait_dscnt 0x0" ::: "memory");
    frag16 pf;
#pragma unroll
    for (int i = 0; i < 8; i++) {
      int kb = (i < 4 ? 2 * i : 16 + 2 * (i - 4)) + 8 * half;
      pf.u[i] = *(const uint32_t*)(pw + l16 * 32 + kb);
    }

    // O += P * V, 8 d-tiles of 16
#pragma unroll
    for (int t = 0; t < 8; t++) {
      frag16 vf;
#pragma unroll
      for (int i = 0; i < 8; i++) {
        int kb = (i < 4 ? 2 * i : 16 + 2 * (i - 4)) + 8 * half;
        vf.u[i] = *(const uint32_t*)(Vs[buf] + (t * 16 + l16) * 32 + kb);
      }
      o[t] = __builtin_amdgcn_wmma_f32_16x16x32_bf16(false, pf.v, false, vf.v,
                                                     (short)0, o[t], false, false);
    }

    // ---- commit next tile to the other buffer
    if (havenext) {
#pragma unroll
      for (int e = 0; e < 16; e++) Vs[nbuf][(sd + e) * 32 + skey] = tv.s[e];
#if !HAVE_TDM
      *(uint4*)(Ks[nbuf] + skey * 128 + sd) = ka0;
      *(uint4*)(Ks[nbuf] + skey * 128 + sd + 8) = ka1;
#else
      if (wave == 0) tdm_wait0();
#endif
    }
    __syncthreads();
  }

#pragma unroll
  for (int i = 0; i < 8; i++) {
    float inv = 1.0f / l_i[i];
    size_t row = (size_t)(b * SEQ + qw + i + 8 * half);
#pragma unroll
    for (int t = 0; t < 8; t++)
      Og[row * HID + h * HEADDIM + t * 16 + l16] = f2bf(o[t][i] * inv);
  }
}

// ------------------------------------------------------------------ launcher
extern "C" void kernel_launch(void* const* d_in, const int* in_sizes, int n_in,
                              void* d_out, int out_size, void* d_ws, size_t ws_size,
                              hipStream_t stream) {
  (void)in_sizes; (void)n_in; (void)out_size; (void)ws_size;
  const float* hidden = (const float*)d_in[0];
  // d_in[1] = attention_mask (causal, computed analytically), d_in[6] = position_ids (arange)
  const float* wq = (const float*)d_in[2];
  const float* wk = (const float*)d_in[3];
  const float* wv = (const float*)d_in[4];
  const float* wo = (const float*)d_in[5];
  float* out = (float*)d_out;

  char* w = (char*)d_ws;
  uint16_t* Xb = (uint16_t*)w; w += (size_t)ROWS * HID * 2;
  uint16_t* Qb = (uint16_t*)w; w += (size_t)ROWS * HID * 2;
  uint16_t* Kb = (uint16_t*)w; w += (size_t)ROWS * KVD * 2;
  uint16_t* Vb = (uint16_t*)w; w += (size_t)ROWS * KVD * 2;
  uint16_t* Ab = (uint16_t*)w;

  int n4 = ROWS * HID / 4;
  k_cvt<<<(n4 + 255) / 256, 256, 0, stream>>>(hidden, Xb, n4);

  dim3 gq(ROWS / 128, HID / 64);
  dim3 gk(ROWS / 128, KVD / 64);
  k_gemm_bf16w<true><<<gq, 256, 0, stream>>>(Xb, wq, Qb, ROWS, HID, HID);
  k_gemm_bf16w<true><<<gk, 256, 0, stream>>>(Xb, wk, Kb, ROWS, KVD, HID);
  k_gemm_bf16w<true><<<gk, 256, 0, stream>>>(Xb, wv, Vb, ROWS, KVD, HID);

  k_rope<<<ROWS * NUMHEADS * 64 / 256, 256, 0, stream>>>(Qb, NUMHEADS, HID);
  k_rope<<<ROWS * NUMKVHEADS * 64 / 256, 256, 0, stream>>>(Kb, NUMKVHEADS, KVD);

  dim3 gf(SEQ / 128, BATCH * NUMHEADS);
  k_flash_attn<<<gf, 256, 0, stream>>>(Qb, Kb, Vb, Ab);

  k_gemm_bf16w<false><<<gq, 256, 0, stream>>>(Ab, wo, out, ROWS, HID, HID);
}
